// RGCNModel_57853209477142
// MI455X (gfx1250) — compile-verified, matrix operand order checked by reading, not run on hardware
//
#include <hip/hip_runtime.h>

#define RREL 4

typedef __attribute__((ext_vector_type(16))) __bf16 v16bf;
typedef __attribute__((ext_vector_type(8)))  float  v8f;
typedef __attribute__((ext_vector_type(4)))  unsigned int u32x4;

union ABfrag { v16bf bf; u32x4 u[2]; };

__device__ __forceinline__ unsigned short f2bf(float f) {
  unsigned int u = __float_as_uint(f);
  unsigned int r = 0x7FFFu + ((u >> 16) & 1u);
  return (unsigned short)((u + r) >> 16);
}

// ---------------- utility kernels ----------------
__global__ void k_zero_f32(float* __restrict__ p, int n) {
  int i = blockIdx.x * blockDim.x + threadIdx.x;
  if (i < n) p[i] = 0.f;
}

__global__ void k_f32_to_bf16(const float* __restrict__ x, unsigned short* __restrict__ o, int n) {
  int i = blockIdx.x * blockDim.x + threadIdx.x;
  if (i < n) o[i] = f2bf(x[i]);
}

__global__ void k_count_edges(const int* __restrict__ dst, const int* __restrict__ et,
                              float* __restrict__ cnt, int E) {
  int e = blockIdx.x * blockDim.x + threadIdx.x;
  if (e < E) unsafeAtomicAdd(&cnt[dst[e] * RREL + et[e]], 1.0f);
}

__global__ void k_make_inv(float* __restrict__ c, int n) {
  int i = blockIdx.x * blockDim.x + threadIdx.x;
  if (i < n) c[i] = 1.0f / fmaxf(c[i], 1.0f);
}

// q = relu(qe @ qn_W + qn_b)   [G,768]x[768,64] -> [G,64]  (tiny)
__global__ void k_q_proj(const float* __restrict__ qe, const float* __restrict__ w,
                         const float* __restrict__ b, float* __restrict__ q, int G) {
  int i = blockIdx.x * blockDim.x + threadIdx.x;
  if (i >= G * 64) return;
  int g = i >> 6, j = i & 63;
  float s = b[j];
  for (int k = 0; k < 768; ++k) s = fmaf(qe[g * 768 + k], w[k * 64 + j], s);
  q[i] = fmaxf(s, 0.f);
}

// combined[n][64+j] = bf16(q[batch[n]][j])
__global__ void k_q_gather(const float* __restrict__ q, const int* __restrict__ batch,
                           unsigned short* __restrict__ comb, int N) {
  int i = blockIdx.x * blockDim.x + threadIdx.x;
  if (i >= N * 64) return;
  int n = i >> 6, j = i & 63;
  comb[(size_t)n * 128 + 64 + j] = f2bf(q[batch[n] * 64 + j]);
}

// ---------------- weight packing into WMMA B-fragment layout ----------------
// Wcat[k][n]: n<out -> root[k][n]; else relation r=n/out-1, col n%out.
// Packed [kt][nt][lane][e] bf16; lane: ncol = nt*16+(lane&15), half=lane>>4;
// element e -> k_local = (e<8 ? e : e+8) + 8*half   (mirror of 16-bit A layout).
__global__ void k_pack_wcat(const float* __restrict__ root, const float* __restrict__ Wrel,
                            int out, int NT, unsigned short* __restrict__ Bpk) {
  int idx = blockIdx.x * blockDim.x + threadIdx.x;
  int total = 4 * NT * 512;                 // KT=4 (K=128, 32/step)
  if (idx >= total) return;
  int e    = idx & 15;
  int lane = (idx >> 4) & 31;
  int nt   = (idx >> 9) % NT;
  int kt   = idx / (NT * 512);
  int n    = nt * 16 + (lane & 15);
  int half = lane >> 4;
  int k    = kt * 32 + ((e < 8) ? e : e + 8) + 8 * half;
  float v;
  if (n < out) {
    v = root[k * out + n];
  } else {
    int r = n / out - 1;
    int c = n - (r + 1) * out;
    v = Wrel[((size_t)r * 128 + k) * out + c];
  }
  Bpk[idx] = f2bf(v);
}

// ---------------- fused WMMA GEMM:  Y[N, OUT5] = A[N,128] @ Wcat ----------------
// Register blocking: 1 M-tile x 4 N-tiles per wave -> A fragment reused 4x,
// 16 v_wmma per wave. OUT5 (= ldy = 5*out) is compile-time: store addressing
// becomes shifts + immediate offsets, no per-row u64 muls.
template <int OUT5>
__global__ __launch_bounds__(128)
void k_rgcn_gemm(const unsigned short* __restrict__ Abf,   // [N x 128] bf16 row-major
                 const unsigned short* __restrict__ Bpk,   // packed [4][NT][32][16]
                 float* __restrict__ Y, int Mtiles, int N) {
  constexpr int NT = OUT5 / 16;
  const int lane = threadIdx.x & 31;
  const int wave = threadIdx.x >> 5;
  const int mt   = blockIdx.x * 4 + wave;
  if (mt >= Mtiles) return;                 // wave-uniform
  const int ntg  = blockIdx.y * 4;          // first of this wave's 4 n-tiles
  const int m    = lane & 15;
  const int half = lane >> 4;

  int arowi = mt * 16 + m;
  if (arowi >= N) arowi = N - 1;            // clamp keeps EXEC all-ones
  const unsigned short* arow = Abf + (size_t)arowi * 128;

  v8f acc0 = {}, acc1 = {}, acc2 = {}, acc3 = {};
#pragma unroll
  for (int kt = 0; kt < 4; ++kt) {
    ABfrag a;
    a.u[0] = *(const u32x4*)(arow + kt * 32 + 8 * half);
    a.u[1] = *(const u32x4*)(arow + kt * 32 + 16 + 8 * half);
    const unsigned short* bb = Bpk + ((size_t)kt * NT + ntg) * 512 + lane * 16;
    ABfrag b0, b1, b2, b3;
    b0.u[0] = *(const u32x4*)(bb);          b0.u[1] = *(const u32x4*)(bb + 8);
    b1.u[0] = *(const u32x4*)(bb + 512);    b1.u[1] = *(const u32x4*)(bb + 520);
    b2.u[0] = *(const u32x4*)(bb + 1024);   b2.u[1] = *(const u32x4*)(bb + 1032);
    b3.u[0] = *(const u32x4*)(bb + 1536);   b3.u[1] = *(const u32x4*)(bb + 1544);
    acc0 = __builtin_amdgcn_wmma_f32_16x16x32_bf16(false, a.bf, false, b0.bf, (short)0, acc0, false, false);
    acc1 = __builtin_amdgcn_wmma_f32_16x16x32_bf16(false, a.bf, false, b1.bf, (short)0, acc1, false, false);
    acc2 = __builtin_amdgcn_wmma_f32_16x16x32_bf16(false, a.bf, false, b2.bf, (short)0, acc2, false, false);
    acc3 = __builtin_amdgcn_wmma_f32_16x16x32_bf16(false, a.bf, false, b3.bf, (short)0, acc3, false, false);
  }

  // C/D layout: VGPR j -> row = mt*16 + j + 8*half, col = lane&15 within tile
  const int ncol = ntg * 16 + m;
  const int row0 = mt * 16 + 8 * half;
  float* yp = Y + (size_t)row0 * OUT5 + ncol;
  if (mt * 16 + 16 <= N) {                  // uniform fast path (full tile)
#pragma unroll
    for (int j = 0; j < 8; ++j) {
      yp[(size_t)j * OUT5 +  0] = acc0[j];
      yp[(size_t)j * OUT5 + 16] = acc1[j];
      yp[(size_t)j * OUT5 + 32] = acc2[j];
      yp[(size_t)j * OUT5 + 48] = acc3[j];
    }
  } else {
#pragma unroll
    for (int j = 0; j < 8; ++j) {
      if (row0 + j < N) {
        yp[(size_t)j * OUT5 +  0] = acc0[j];
        yp[(size_t)j * OUT5 + 16] = acc1[j];
        yp[(size_t)j * OUT5 + 32] = acc2[j];
        yp[(size_t)j * OUT5 + 48] = acc3[j];
      }
    }
  }
}

// ------- edge scatter: Y[dst][0..out) += inv[dst][r] * Y[src][(1+r)*out ..) -------
__global__ void k_scatter(const int* __restrict__ src, const int* __restrict__ dst,
                          const int* __restrict__ et, const float* __restrict__ inv,
                          float* __restrict__ Y, int ldy, int out, int E) {
  const int cpe = out >> 2;                                   // float4 chunks per edge
  long gid = (long)blockIdx.x * blockDim.x + threadIdx.x;
  long e = gid / cpe;
  if (e >= E) return;
  int c = (int)(gid - e * cpe) * 4;
  int r = et[e], s = src[e], d = dst[e];
  float w = inv[d * RREL + r];
  const float* hp = Y + (size_t)s * ldy + (size_t)(r + 1) * out + c;
  float h0 = hp[0], h1 = hp[1], h2 = hp[2], h3 = hp[3];
  float* op = Y + (size_t)d * ldy + c;
  unsafeAtomicAdd(op + 0, h0 * w);
  unsafeAtomicAdd(op + 1, h1 * w);
  unsafeAtomicAdd(op + 2, h2 * w);
  unsafeAtomicAdd(op + 3, h3 * w);
}

// ---------------- epilogues ----------------
__global__ void k_relu_bias_bf16(const float* __restrict__ Y, int ldy,
                                 const float* __restrict__ bias, int out,
                                 unsigned short* __restrict__ dstbf, int N) {
  int i = blockIdx.x * blockDim.x + threadIdx.x;
  if (i >= N * out) return;
  int n = i / out, c = i - n * out;
  float v = Y[(size_t)n * ldy + c] + bias[c];
  dstbf[(size_t)n * 128 + c] = f2bf(fmaxf(v, 0.f));
}

__global__ void k_bias_final(const float* __restrict__ Y, int ldy,
                             const float* __restrict__ bias, int out,
                             float* __restrict__ o, int N) {
  int i = blockIdx.x * blockDim.x + threadIdx.x;
  if (i >= N * out) return;
  int n = i / out, c = i - n * out;
  o[(size_t)n * out + c] = Y[(size_t)n * ldy + c] + bias[c];
}

// ---------------- host ----------------
extern "C" void kernel_launch(void* const* d_in, const int* in_sizes, int n_in,
                              void* d_out, int out_size, void* d_ws, size_t ws_size,
                              hipStream_t stream) {
  const float* x     = (const float*)d_in[0];
  const int*   ei    = (const int*)d_in[1];
  const int*   ea    = (const int*)d_in[2];
  const int*   batch = (const int*)d_in[3];
  const float* qe    = (const float*)d_in[4];
  const float* qnW   = (const float*)d_in[5];
  const float* qnb   = (const float*)d_in[6];
  const float* W0 = (const float*)d_in[7];  const float* r0 = (const float*)d_in[8];  const float* b0 = (const float*)d_in[9];
  const float* W1 = (const float*)d_in[10]; const float* r1 = (const float*)d_in[11]; const float* b1 = (const float*)d_in[12];
  const float* W2 = (const float*)d_in[13]; const float* r2 = (const float*)d_in[14]; const float* b2 = (const float*)d_in[15];
  const float* W3 = (const float*)d_in[16]; const float* r3 = (const float*)d_in[17]; const float* b3 = (const float*)d_in[18];

  const int N = in_sizes[3];
  const int E = in_sizes[2];
  const int G = in_sizes[4] / 768;
  const int* src = ei;
  const int* dst = ei + E;

  // workspace partition (256B aligned regions)
  char* wsp = (char*)d_ws;
  auto alloc = [&](size_t bytes) -> char* {
    char* p = wsp;
    wsp += (bytes + 255) & ~(size_t)255;
    return p;
  };
  float*          inv  = (float*)alloc((size_t)N * RREL * 4);
  float*          qbuf = (float*)alloc((size_t)G * 64 * 4);
  unsigned short* xbf  = (unsigned short*)alloc((size_t)N * 128 * 2);  // ping
  unsigned short* cbf  = (unsigned short*)alloc((size_t)N * 128 * 2);  // pong
  float*          Y    = (float*)alloc((size_t)N * 640 * 4);
  unsigned short* Bpk0 = (unsigned short*)alloc((size_t)4 * 20 * 512 * 2);
  unsigned short* Bpk1 = (unsigned short*)alloc((size_t)4 * 40 * 512 * 2);
  unsigned short* Bpk2 = (unsigned short*)alloc((size_t)4 * 40 * 512 * 2);
  unsigned short* Bpk3 = (unsigned short*)alloc((size_t)4 * 20 * 512 * 2);

  const int TB = 256;
  auto nb = [&](long n) { return (int)((n + TB - 1) / TB); };
  const int Mtiles = (N + 15) / 16;
  const int Mblk   = (Mtiles + 3) / 4;      // 4 m-tiles (waves) per block

  // 1) per-(dst, relation) inverse counts (shared by all 4 layers)
  k_zero_f32<<<nb((long)N * RREL), TB, 0, stream>>>(inv, N * RREL);
  k_count_edges<<<nb(E), TB, 0, stream>>>(dst, ea, inv, E);
  k_make_inv<<<nb((long)N * RREL), TB, 0, stream>>>(inv, N * RREL);

  // 2) activations -> bf16; question projection
  k_f32_to_bf16<<<nb((long)N * 128), TB, 0, stream>>>(x, xbf, N * 128);
  k_q_proj<<<nb((long)G * 64), TB, 0, stream>>>(qe, qnW, qnb, qbuf, G);

  // 3) pack all layer weights into WMMA B-fragment layout
  k_pack_wcat<<<nb(4L * 20 * 512), TB, 0, stream>>>(r0, W0, 64,  20, Bpk0);
  k_pack_wcat<<<nb(4L * 40 * 512), TB, 0, stream>>>(r1, W1, 128, 40, Bpk1);
  k_pack_wcat<<<nb(4L * 40 * 512), TB, 0, stream>>>(r2, W2, 128, 40, Bpk2);
  k_pack_wcat<<<nb(4L * 20 * 512), TB, 0, stream>>>(r3, W3, 64,  20, Bpk3);

  // ----- layer 0: in=128, out=64, Y is [N,320] -----
  k_rgcn_gemm<320><<<dim3(Mblk, 5), 128, 0, stream>>>(xbf, Bpk0, Y, Mtiles, N);
  k_scatter<<<nb((long)E * 16), TB, 0, stream>>>(src, dst, ea, inv, Y, 320, 64, E);
  k_relu_bias_bf16<<<nb((long)N * 64), TB, 0, stream>>>(Y, 320, b0, 64, cbf, N);
  k_q_gather<<<nb((long)N * 64), TB, 0, stream>>>(qbuf, batch, cbf, N);   // cols 64..127

  // ----- layer 1: in=128, out=128, Y is [N,640] -----
  k_rgcn_gemm<640><<<dim3(Mblk, 10), 128, 0, stream>>>(cbf, Bpk1, Y, Mtiles, N);
  k_scatter<<<nb((long)E * 32), TB, 0, stream>>>(src, dst, ea, inv, Y, 640, 128, E);
  k_relu_bias_bf16<<<nb((long)N * 128), TB, 0, stream>>>(Y, 640, b1, 128, xbf, N);

  // ----- layer 2: in=128, out=128 -----
  k_rgcn_gemm<640><<<dim3(Mblk, 10), 128, 0, stream>>>(xbf, Bpk2, Y, Mtiles, N);
  k_scatter<<<nb((long)E * 32), TB, 0, stream>>>(src, dst, ea, inv, Y, 640, 128, E);
  k_relu_bias_bf16<<<nb((long)N * 128), TB, 0, stream>>>(Y, 640, b2, 128, cbf, N);

  // ----- layer 3: in=128, out=64, no relu, fp32 out -----
  k_rgcn_gemm<320><<<dim3(Mblk, 5), 128, 0, stream>>>(cbf, Bpk3, Y, Mtiles, N);
  k_scatter<<<nb((long)E * 16), TB, 0, stream>>>(src, dst, ea, inv, Y, 320, 64, E);
  k_bias_final<<<nb((long)N * 64), TB, 0, stream>>>(Y, 320, b3, 64, (float*)d_out, N);
}
